// CausalSelfAttention_44195213475965
// MI455X (gfx1250) — matrix operand order repaired and checked by reading, not compiled, not run on hardware
//
#include <hip/hip_runtime.h>
#include <hip/hip_bf16.h>

// ---------------------------------------------------------------------------
// CausalSelfAttention for MI455X (gfx1250, wave32)
//   Pass 0: convert x -> bf16; transpose+convert w_attn, w_proj -> bf16 [N][K]
//   Pass 1: qkv = x @ w_attn   (bf16 WMMA GEMM, async-LDS double-buffered)
//   Pass 2: flash attention    (bf16 WMMA QK^T / PV, f32 online softmax,
//                               DPP row-rotation reductions)
//   Pass 3: out = attn @ w_proj
// B=4, T=2048, C=1024, H=16, D=64
// ---------------------------------------------------------------------------

typedef __attribute__((ext_vector_type(16))) __bf16 v16bf;
typedef __attribute__((ext_vector_type(8)))  __bf16 v8bf;
typedef __attribute__((ext_vector_type(4)))  __bf16 v4bf;
typedef __attribute__((ext_vector_type(8)))  float  v8f;
typedef int v4i __attribute__((vector_size(16)));   // matches builtin param type

#define WMMA_BF16(a, b, c) \
  __builtin_amdgcn_wmma_f32_16x16x32_bf16(false, (a), false, (b), (short)0, (c), false, false)

#if __has_builtin(__builtin_amdgcn_global_load_async_to_lds_b128)
#define HAVE_ASYNC 1
#else
#define HAVE_ASYNC 0
#endif

// 16-byte global -> LDS copy (async DMA when available, sync fallback otherwise)
static __device__ __forceinline__ void copy16(__bf16* dst_lds, const __bf16* src_glb) {
#if HAVE_ASYNC
  __builtin_amdgcn_global_load_async_to_lds_b128(
      (__attribute__((address_space(1))) v4i*)(src_glb),
      (__attribute__((address_space(3))) v4i*)(dst_lds), 0, 0);
#else
  *(uint4*)dst_lds = *(const uint4*)src_glb;
#endif
}

static __device__ __forceinline__ void wait_async0() {
#if HAVE_ASYNC
#if __has_builtin(__builtin_amdgcn_s_wait_asynccnt)
  __builtin_amdgcn_s_wait_asynccnt(0);
#else
  asm volatile("s_wait_asynccnt 0x0" ::: "memory");
#endif
#endif
}

static __device__ __forceinline__ void wait_async8() {
#if HAVE_ASYNC
#if __has_builtin(__builtin_amdgcn_s_wait_asynccnt)
  __builtin_amdgcn_s_wait_asynccnt(8);
#else
  asm volatile("s_wait_asynccnt 0x8" ::: "memory");
#endif
#endif
}

// DPP16 row rotate (within 16-lane row): VALU-only cross-lane data movement.
// CTRL 0x121/0x122/0x124/0x128 = row_ror:1/2/4/8.
template <int CTRL>
static __device__ __forceinline__ float dpp_rotf(float x) {
  return __int_as_float(
      __builtin_amdgcn_update_dpp(0, __float_as_int(x), CTRL, 0xf, 0xf, true));
}

// max / sum across the 16-lane half (cyclic rotate-combine over 1,2,4,8)
static __device__ __forceinline__ float rowmax16(float x) {
  x = fmaxf(x, dpp_rotf<0x121>(x));
  x = fmaxf(x, dpp_rotf<0x122>(x));
  x = fmaxf(x, dpp_rotf<0x124>(x));
  x = fmaxf(x, dpp_rotf<0x128>(x));
  return x;
}
static __device__ __forceinline__ float rowsum16(float x) {
  x += dpp_rotf<0x121>(x);
  x += dpp_rotf<0x122>(x);
  x += dpp_rotf<0x124>(x);
  x += dpp_rotf<0x128>(x);
  return x;
}

static __device__ __forceinline__ void cvt4(const float4 v, __bf16* q) {
  q[0] = (__bf16)v.x; q[1] = (__bf16)v.y; q[2] = (__bf16)v.z; q[3] = (__bf16)v.w;
}

// A-matrix fragment (16 rows x 32 K), LDS layout [row][k], `stride` elems/row.
// ISA: lanes 0-15 = rows, K {0..7,16..23}; lanes 16-31 = rows, K {8..15,24..31}.
static __device__ __forceinline__ v16bf load_frag_a(const __bf16* p, int row0,
                                                    int coloff, int stride) {
  const int lane = threadIdx.x & 31;
  const __bf16* base = p + (row0 + (lane & 15)) * stride + coloff + ((lane >> 4) * 8);
  v8bf lo = *(const v8bf*)(base);
  v8bf hi = *(const v8bf*)(base + 16);
  v16bf f;
#pragma unroll
  for (int i = 0; i < 8; ++i) { f[i] = lo[i]; f[i + 8] = hi[i]; }
  return f;
}

// B-matrix fragment (32 K x 16 cols), LDS layout [col][k], `stride` elems/row.
// ISA: lanes 0-15 = cols, K {0..7,16..23}; lanes 16-31 = cols, K {8..15,24..31}.
static __device__ __forceinline__ v16bf load_frag_b(const __bf16* p, int row0,
                                                    int coloff, int stride) {
  const int lane = threadIdx.x & 31;
  const __bf16* base = p + (row0 + (lane & 15)) * stride + coloff + ((lane >> 4) * 8);
  v8bf lo = *(const v8bf*)(base);
  v8bf hi = *(const v8bf*)(base + 16);
  v16bf f;
#pragma unroll
  for (int i = 0; i < 8; ++i) { f[i] = lo[i]; f[i + 8] = hi[i]; }
  return f;
}

// ---------------------------------------------------------------------------
// Pass 0 kernels: one-time precision / layout conversion
// ---------------------------------------------------------------------------
__global__ __launch_bounds__(256)
void cvt_bf16(const float* __restrict__ src, __bf16* __restrict__ dst) {
  const size_t idx = (size_t)blockIdx.x * 256 + threadIdx.x;   // one float4 each
  float4 v = ((const float4*)src)[idx];
  v4bf q; cvt4(v, (__bf16*)&q);
  ((v4bf*)dst)[idx] = q;
}

// src [K][N] fp32 row-major -> dst [N][K] bf16 (pre-transposed weights)
template <int K, int N>
__global__ __launch_bounds__(256)
void transpose_cvt(const float* __restrict__ src, __bf16* __restrict__ dst) {
  const int idx = blockIdx.x * 256 + threadIdx.x;  // over N * K/4
  const int n = idx % N;
  const int k4 = (idx / N) * 4;
  v4bf q;
#pragma unroll
  for (int j = 0; j < 4; ++j) q[j] = (__bf16)src[(size_t)(k4 + j) * N + n];
  *(v4bf*)(dst + (size_t)n * K + k4) = q;
}

// ---------------------------------------------------------------------------
// bf16 GEMM: C[M,N] = A[M,K] * Bt[N,K]^T.  128x128 block, 4 waves of 64x64,
// K step 32, async global->LDS double buffering.
// ---------------------------------------------------------------------------
#define GSTR 40  // LDS row stride (bf16 elems) for 32-wide K tiles: 80B, bank-friendly

template <int N, int K, bool OUT_BF16>
__global__ __launch_bounds__(128)
void gemm_bf16_wmma(const __bf16* __restrict__ A, const __bf16* __restrict__ Bt,
                    void* __restrict__ Cout) {
  __shared__ __align__(16) __bf16 sA[2][128 * GSTR];
  __shared__ __align__(16) __bf16 sB[2][128 * GSTR];

  const int tid  = threadIdx.x;
  const int lane = tid & 31;
  const int wave = tid >> 5;
  const int wm = wave >> 1, wn = wave & 1;
  const int half = lane >> 4, col = lane & 15;
  const int M0 = blockIdx.y * 128;
  const int N0 = blockIdx.x * 128;

  v8f acc[16];
#pragma unroll
  for (int i = 0; i < 16; ++i)
#pragma unroll
    for (int r = 0; r < 8; ++r) acc[i][r] = 0.0f;

  // stage one 128x32 A tile + 128x32 B tile: 8 x 16B chunks per thread
  auto stage = [&](int buf, int k0) {
#pragma unroll
    for (int i = 0; i < 4; ++i) {
      const int f = tid + i * 128;          // 512 chunks per matrix
      const int row = f >> 2, c = (f & 3) * 8;
      copy16(&sA[buf][row * GSTR + c], A  + (size_t)(M0 + row) * K + k0 + c);
      copy16(&sB[buf][row * GSTR + c], Bt + (size_t)(N0 + row) * K + k0 + c);
    }
  };

  constexpr int NK = K / 32;
  stage(0, 0);
  int cur = 0;
  for (int kt = 0; kt < NK; ++kt) {
    __syncthreads();                         // buf[cur^1] fully consumed
    if (kt + 1 < NK) { stage(cur ^ 1, (kt + 1) * 32); wait_async8(); }
    else             { wait_async0(); }
    __syncthreads();                         // buf[cur] visible to all waves

    const __bf16* pA = sA[cur];
    const __bf16* pB = sB[cur];
    v16bf af[4], bfr[4];
#pragma unroll
    for (int mi = 0; mi < 4; ++mi) af[mi]  = load_frag_a(pA, wm * 64 + mi * 16, 0, GSTR);
#pragma unroll
    for (int ni = 0; ni < 4; ++ni) bfr[ni] = load_frag_b(pB, wn * 64 + ni * 16, 0, GSTR);
#pragma unroll
    for (int mi = 0; mi < 4; ++mi)
#pragma unroll
      for (int ni = 0; ni < 4; ++ni)
        acc[mi * 4 + ni] = WMMA_BF16(af[mi], bfr[ni], acc[mi * 4 + ni]);
    cur ^= 1;
  }

  // epilogue: C layout — VGPR r: lanes 0-15 -> M=r, lanes 16-31 -> M=8+r
#pragma unroll
  for (int mi = 0; mi < 4; ++mi)
#pragma unroll
    for (int ni = 0; ni < 4; ++ni)
#pragma unroll
      for (int r = 0; r < 8; ++r) {
        const int gm = M0 + wm * 64 + mi * 16 + half * 8 + r;
        const int gn = N0 + wn * 64 + ni * 16 + col;
        if constexpr (OUT_BF16)
          ((__bf16*)Cout)[(size_t)gm * N + gn] = (__bf16)acc[mi * 4 + ni][r];
        else
          ((float*)Cout)[(size_t)gm * N + gn] = acc[mi * 4 + ni][r];
      }
}

// ---------------------------------------------------------------------------
// Flash attention: block per (64-row q-tile, head, batch); 4 waves x 16 q rows.
// qkv is bf16 [B*T][3C]; output bf16 [B*T][C]. scale = 1/sqrt(C) = 1/32.
// ---------------------------------------------------------------------------
#define QSTR 72  // LDS row stride for 64-wide d rows: 144B (multiple of 16B)

__global__ __launch_bounds__(128)
void attn_flash_wmma(const __bf16* __restrict__ qkv, __bf16* __restrict__ outp) {
  constexpr int kT = 2048, kC = 1024, k3C = 3072;
  __shared__ __align__(16) __bf16 sQ[64 * QSTR];
  __shared__ __align__(16) __bf16 sK[32 * QSTR];
  __shared__ __align__(16) __bf16 sV[64 * GSTR];        // V transposed: [d][key]
  __shared__ __align__(16) __bf16 sP[4 * 16 * GSTR];

  const int tid  = threadIdx.x;
  const int lane = tid & 31;
  const int wave = tid >> 5;
  const int half = lane >> 4, col = lane & 15;
  const int q0 = blockIdx.x * 64;
  const int h  = blockIdx.y;
  const int b  = blockIdx.z;
  const size_t baseRow = (size_t)b * kT;
  const int hoff = h * 64;
  const float scale = 0.03125f;  // 1/sqrt(1024)

  // ---- stage Q (64 q x 64 d): 512 chunks of 16B, 4 per thread ----
#pragma unroll
  for (int i = 0; i < 4; ++i) {
    const int f = tid + i * 128;
    const int row = f >> 3, c = (f & 7) * 8;
    copy16(&sQ[row * QSTR + c], qkv + (baseRow + q0 + row) * k3C + hoff + c);
  }
  wait_async0();
  __syncthreads();

  const v16bf qf0 = load_frag_a(sQ, wave * 16, 0,  QSTR);  // d 0..31
  const v16bf qf1 = load_frag_a(sQ, wave * 16, 32, QSTR);  // d 32..63

  v8f o[4];
  float mrow[8], lrow[8];
#pragma unroll
  for (int j = 0; j < 4; ++j)
#pragma unroll
    for (int r = 0; r < 8; ++r) o[j][r] = 0.0f;
#pragma unroll
  for (int r = 0; r < 8; ++r) { mrow[r] = -3.0e38f; lrow[r] = 0.0f; }

  const int qrowBase = q0 + wave * 16 + half * 8;
  const int nKT = blockIdx.x * 2 + 2;  // causal: keys < q0+64

  for (int kt = 0; kt < nKT; ++kt) {
    const int t0 = kt * 32;
    // ---- stage K (32 keys x 64 d) via async: 256 chunks, 2 per thread ----
#pragma unroll
    for (int i = 0; i < 2; ++i) {
      const int f = tid + i * 128;
      const int row = f >> 3, c = (f & 7) * 8;
      copy16(&sK[row * QSTR + c], qkv + (baseRow + t0 + row) * k3C + kC + hoff + c);
    }
    // ---- stage V transposed [d][key]: thread = (d-quad, key-quad) ----
    {
      const int dq = (tid >> 3) * 4, kq = (tid & 7) * 4;
      const __bf16* vb = qkv + (baseRow + t0 + kq) * k3C + 2 * kC + hoff + dq;
      v4bf l0 = *(const v4bf*)(vb);
      v4bf l1 = *(const v4bf*)(vb + k3C);
      v4bf l2 = *(const v4bf*)(vb + 2 * k3C);
      v4bf l3 = *(const v4bf*)(vb + 3 * k3C);
#pragma unroll
      for (int j = 0; j < 4; ++j) {
        v4bf q; q[0] = l0[j]; q[1] = l1[j]; q[2] = l2[j]; q[3] = l3[j];
        *(v4bf*)(sV + (dq + j) * GSTR + kq) = q;
      }
    }
    wait_async0();
    __syncthreads();

    // ---- S = Q K^T : two 16x16 tiles (keys t0..+15, t0+16..+31) ----
    v8f s0, s1;
#pragma unroll
    for (int r = 0; r < 8; ++r) { s0[r] = 0.0f; s1[r] = 0.0f; }
    {
      v16bf kb00 = load_frag_b(sK, 0,  0,  QSTR);
      v16bf kb01 = load_frag_b(sK, 0,  32, QSTR);
      v16bf kb10 = load_frag_b(sK, 16, 0,  QSTR);
      v16bf kb11 = load_frag_b(sK, 16, 32, QSTR);
      s0 = WMMA_BF16(qf0, kb00, s0);
      s0 = WMMA_BF16(qf1, kb01, s0);
      s1 = WMMA_BF16(qf0, kb10, s1);
      s1 = WMMA_BF16(qf1, kb11, s1);
    }

    // ---- online softmax (row stats via DPP row rotations, VALU-only) ----
    float p0[8], p1[8];
#pragma unroll
    for (int r = 0; r < 8; ++r) {
      const int qg = qrowBase + r;
      float a0 = (t0 + col      <= qg) ? s0[r] * scale : -3.0e38f;
      float a1 = (t0 + 16 + col <= qg) ? s1[r] * scale : -3.0e38f;
      const float mx = rowmax16(fmaxf(a0, a1));
      const float mn = fmaxf(mrow[r], mx);
      const float e0 = __expf(a0 - mn), e1 = __expf(a1 - mn);
      const float sum = rowsum16(e0 + e1);
      const float alpha = __expf(mrow[r] - mn);
      lrow[r] = lrow[r] * alpha + sum;
      mrow[r] = mn;
      p0[r] = e0; p1[r] = e1;
#pragma unroll
      for (int j = 0; j < 4; ++j) o[j][r] *= alpha;
    }

    // ---- P (f32, C-layout) -> LDS bf16 [q][key], reload as A fragment ----
    __bf16* myP = sP + wave * 16 * GSTR;
#pragma unroll
    for (int r = 0; r < 8; ++r) {
      myP[(half * 8 + r) * GSTR + col]      = (__bf16)p0[r];
      myP[(half * 8 + r) * GSTR + 16 + col] = (__bf16)p1[r];
    }
    __syncthreads();

    {
      v16bf pf = load_frag_a(myP, 0, 0, GSTR);
#pragma unroll
      for (int j = 0; j < 4; ++j) {
        v16bf vbf = load_frag_b(sV, j * 16, 0, GSTR);
        o[j] = WMMA_BF16(pf, vbf, o[j]);
      }
    }
    __syncthreads();  // K/V/P fully consumed before next stage
  }

  // ---- normalize and write attention output (bf16) ----
#pragma unroll
  for (int r = 0; r < 8; ++r) {
    const float inv = 1.0f / lrow[r];
    const size_t row = baseRow + (size_t)(qrowBase + r);
#pragma unroll
    for (int j = 0; j < 4; ++j)
      outp[row * kC + hoff + j * 16 + col] = (__bf16)(o[j][r] * inv);
  }
}

// ---------------------------------------------------------------------------
extern "C" void kernel_launch(void* const* d_in, const int* in_sizes, int n_in,
                              void* d_out, int out_size, void* d_ws, size_t ws_size,
                              hipStream_t stream) {
  (void)in_sizes; (void)n_in; (void)out_size; (void)ws_size;
  constexpr int Bz = 4, T = 2048, C = 1024, H = 16;
  constexpr int M = Bz * T;  // 8192

  const float* x      = (const float*)d_in[0];
  const float* w_attn = (const float*)d_in[1];
  const float* w_proj = (const float*)d_in[2];

  __bf16* xb   = (__bf16*)d_ws;                    // 16 MB
  __bf16* waT  = xb  + (size_t)M * C;              //  6 MB  [3C][C]
  __bf16* wpT  = waT + (size_t)3 * C * C;          //  2 MB  [C][C]
  __bf16* qkv  = wpT + (size_t)C * C;              // 48 MB  [M][3C]
  __bf16* attn = qkv + (size_t)M * 3 * C;          // 16 MB  [M][C]

  // Pass 0: precision / layout conversion (one-time)
  cvt_bf16<<<(M * C) / (4 * 256), 256, 0, stream>>>(x, xb);
  transpose_cvt<C, 3 * C><<<(3 * C * C) / (4 * 256), 256, 0, stream>>>(w_attn, waT);
  transpose_cvt<C, C><<<(C * C) / (4 * 256), 256, 0, stream>>>(w_proj, wpT);
  // Pass 1: qkv = x @ w_attn
  gemm_bf16_wmma<3 * C, C, true><<<dim3((3 * C) / 128, M / 128), 128, 0, stream>>>(xb, waT, qkv);
  // Pass 2: flash attention
  attn_flash_wmma<<<dim3(T / 64, H, Bz), 128, 0, stream>>>(qkv, attn);
  // Pass 3: out = attn @ w_proj
  gemm_bf16_wmma<C, C, false><<<dim3(C / 128, M / 128), 128, 0, stream>>>(attn, wpT, d_out);
}
